// SelfAttentionPose_27530740367859
// MI455X (gfx1250) — matrix operand order repaired and checked by reading, not compiled, not run on hardware
//
#include <hip/hip_runtime.h>

// ---------------------------------------------------------------------------
// SelfAttentionPose for MI455X (gfx1250, wave32, WMMA).
// f32 in/out, f16 WMMA with f32 accumulation (v_wmma_f32_16x16x32_f16).
// Weights pre-transposed/converted to f16 so every B-fragment is 2x b128.
// K chunks staged LDS-direct via global_load_async_to_lds_b128 (ASYNCcnt).
// ---------------------------------------------------------------------------

#define NI    512
#define SEQ   1024
#define NHD   8
#define DH    64
#define PDIM  128
#define KDIM2 514          // 512 + 2 pose channels
#define KPAD  544          // 17 * 32, zero-padded K for the k/v projections
#define EPSLN 1e-5f

typedef __attribute__((ext_vector_type(16))) _Float16 v16h;
typedef __attribute__((ext_vector_type(8)))  float    v8f;

union AB16 { v16h v; _Float16 h[16]; uint4 q[2]; };

__device__ __forceinline__ v8f wmma32(v16h a, v16h b, v8f c) {
  // D = A(16x32 f16) * B(32x16 f16) + C(16x16 f32)
  return __builtin_amdgcn_wmma_f32_16x16x32_f16(false, a, false, b,
                                                (short)0, c, false, false);
}

// A-matrix 16x32 f16 layout (ISA 7.12.2): lane m=row; element e<8 -> K=k0+hi*8+e,
// e>=8 -> K=k0+16+hi*8+(e-8). Both runs are 8 contiguous halves -> two b128 loads.
__device__ __forceinline__ v16h load_afrag(const _Float16* row, int k0, int hi) {
  AB16 f;
  f.q[0] = *(const uint4*)(row + k0 + hi * 8);
  f.q[1] = *(const uint4*)(row + k0 + 16 + hi * 8);
  return f.v;
}

// B-matrix 32x16 f16 layout: lane = column N, element e<8 -> K = hi*8 + e,
// e>=8 -> K = 16 + hi*8 + (e-8) (same K order as the A fragment).
__device__ __forceinline__ v16h load_bfrag(const _Float16* src, int hi) {
  AB16 f;
  f.q[0] = *(const uint4*)(src + hi * 8);
  f.q[1] = *(const uint4*)(src + 16 + hi * 8);
  return f.v;
}

// Async LDS-direct copy (CDNA5): per-lane 16B global -> LDS, tracked by ASYNCcnt.
__device__ __forceinline__ void async_ld_b128(unsigned lds_addr, const void* g) {
  asm volatile("global_load_async_to_lds_b128 %0, %1, off"
               :: "v"(lds_addr), "v"((unsigned long long)(uintptr_t)g)
               : "memory");
}
__device__ __forceinline__ void wait_async0() {
  asm volatile("s_wait_asynccnt 0x0" ::: "memory");
}

// Row reductions for the C-layout: one matrix row lives in 16 contiguous lanes
// (lanes 0-15 for rows 0-7, lanes 16-31 for rows 8-15) -> width-16 shuffles.
__device__ __forceinline__ float redmax16(float v) {
#pragma unroll
  for (int off = 1; off < 16; off <<= 1) v = fmaxf(v, __shfl_xor(v, off, 16));
  return v;
}
__device__ __forceinline__ float redsum16(float v) {
#pragma unroll
  for (int off = 1; off < 16; off <<= 1) v += __shfl_xor(v, off, 16);
  return v;
}

// ---------------------------------------------------------------------------
// Kernel 0: weight prep. Wt[col][k] = (f16) W[k][col], K zero-padded to kpad.
// ---------------------------------------------------------------------------
__global__ __launch_bounds__(256) void wprep_kernel(
    const float* __restrict__ W, _Float16* __restrict__ Wt,
    int kdim, int kpad) {
  int idx = blockIdx.x * 256 + threadIdx.x;   // col * kpad + k
  int col = idx / kpad, k = idx - col * kpad;
  Wt[idx] = (_Float16)((k < kdim) ? W[k * NI + col] : 0.f);
}

// ---------------------------------------------------------------------------
// Kernel 1: pe[n][j][i] = pose_embed[n][i][:] . pose_W[:][j] + pose_b[j]
// ---------------------------------------------------------------------------
__global__ __launch_bounds__(256) void pose_kernel(
    const float* __restrict__ pose_embed, const float* __restrict__ pose_W,
    const float* __restrict__ pose_b, float* __restrict__ pe_t) {
  int idx = blockIdx.x * 256 + threadIdx.x;   // 8 * 1024
  int n = idx >> 10, j = idx & 1023;
  const float* e0 = pose_embed + (size_t)(n * 2 + 0) * PDIM;
  const float* e1 = pose_embed + (size_t)(n * 2 + 1) * PDIM;
  float a0 = pose_b[j], a1 = pose_b[j];
#pragma unroll 4
  for (int d = 0; d < PDIM; ++d) {
    float w = pose_W[d * SEQ + j];
    a0 += e0[d] * w;
    a1 += e1[d] * w;
  }
  pe_t[(size_t)idx * 2 + 0] = a0;
  pe_t[(size_t)idx * 2 + 1] = a1;
}

// ---------------------------------------------------------------------------
// One 16x16 output tile set of a projection GEMM. NCHUNK/KP are compile-time
// so the K loop fully unrolls (no vector loop-mask bookkeeping around WMMA).
// ---------------------------------------------------------------------------
template <int NCHUNK, int KP>
__device__ __forceinline__ void proj_tiles(
    const _Float16 (*__restrict__ A)[KPAD], const _Float16* __restrict__ Wt,
    const float* __restrict__ Bv, _Float16* __restrict__ dst,
    int n, int s0, int wv, int m, int hi) {
  for (int t = wv; t < 32; t += 8) {
    int col = t * 16 + m;
    const _Float16* wcol = Wt + (size_t)col * KP;
    v8f acc;
#pragma unroll
    for (int r = 0; r < 8; ++r) acc[r] = 0.f;
#pragma unroll
    for (int kc = 0; kc < NCHUNK; ++kc) {
      int k0 = kc * 32;
      v16h a = load_afrag(&A[m][0], k0, hi);
      v16h b = load_bfrag(wcol + k0, hi);
      acc = wmma32(a, b, acc);
    }
    float bias = Bv[col];
    int hh = col >> 6, d = col & 63;
#pragma unroll
    for (int r = 0; r < 8; ++r) {
      int R = r + hi * 8;
      dst[(((size_t)n * NHD + hh) * SEQ + s0 + R) * DH + d] =
          (_Float16)(acc[r] + bias);
    }
  }
}

// ---------------------------------------------------------------------------
// Kernel 2: per (n, 16-row seq tile): layernorm(x) and layernorm([x,pe]),
// then Q/K/V projections via WMMA. Output f16 [n][h][s][64].
// ---------------------------------------------------------------------------
__global__ __launch_bounds__(256) void qkv_kernel(
    const float* __restrict__ x, const float* __restrict__ pe_t,
    const float* __restrict__ ng, const float* __restrict__ nb,
    const float* __restrict__ png, const float* __restrict__ pnb,
    const _Float16* __restrict__ qWt, const float* __restrict__ qB,
    const _Float16* __restrict__ kWt, const float* __restrict__ kB,
    const _Float16* __restrict__ vWt, const float* __restrict__ vB,
    _Float16* __restrict__ qh, _Float16* __restrict__ kh,
    _Float16* __restrict__ vh) {
  int n = blockIdx.x >> 6;
  int s0 = (blockIdx.x & 63) * 16;

  __shared__ _Float16 xt[16][NI]    __attribute__((aligned(16)));  // 16 KB
  __shared__ _Float16 xnA[16][KPAD] __attribute__((aligned(16)));  // 17 KB
  __shared__ _Float16 xpA[16][KPAD] __attribute__((aligned(16)));  // 17 KB

  // Stage x tile (xs[s][c] = x[n][c][s]) as f16.
  for (int i = threadIdx.x; i < 16 * NI; i += 256) {
    int r = i & 15, c = i >> 4;
    xt[r][c] = (_Float16)x[((size_t)n * NI + c) * SEQ + s0 + r];
  }
  __syncthreads();

  // Layernorms: threads 0-15 -> xn rows (512 ch); 16-31 -> xpn rows (514 ch).
  if (threadIdx.x < 32) {
    int row = threadIdx.x & 15;
    bool pose = threadIdx.x >= 16;
    float pe0 = pe_t[((size_t)n * SEQ + s0 + row) * 2 + 0];
    float pe1 = pe_t[((size_t)n * SEQ + s0 + row) * 2 + 1];
    int cnt = pose ? KDIM2 : NI;
    float sum = 0.f;
    for (int c = 0; c < NI; ++c) sum += (float)xt[row][c];
    if (pose) sum += pe0 + pe1;
    float mean = sum / (float)cnt;
    float var = 0.f;
    for (int c = 0; c < NI; ++c) {
      float d = (float)xt[row][c] - mean;
      var += d * d;
    }
    if (pose) { float d0 = pe0 - mean, d1 = pe1 - mean; var += d0 * d0 + d1 * d1; }
    float inv = rsqrtf(var / (float)cnt + EPSLN);
    if (!pose) {
      for (int c = 0; c < NI; ++c)
        xnA[row][c] = (_Float16)(((float)xt[row][c] - mean) * inv * ng[c] + nb[c]);
      for (int c = NI; c < KPAD; ++c) xnA[row][c] = (_Float16)0.f;
    } else {
      for (int c = 0; c < NI; ++c)
        xpA[row][c] = (_Float16)(((float)xt[row][c] - mean) * inv * png[c] + pnb[c]);
      xpA[row][NI + 0] = (_Float16)((pe0 - mean) * inv * png[NI + 0] + pnb[NI + 0]);
      xpA[row][NI + 1] = (_Float16)((pe1 - mean) * inv * png[NI + 1] + pnb[NI + 1]);
      for (int c = KDIM2; c < KPAD; ++c) xpA[row][c] = (_Float16)0.f;
    }
  }
  __syncthreads();

  int lane = threadIdx.x & 31, wv = threadIdx.x >> 5;
  int m = lane & 15, hi = lane >> 4;

  // Three GEMMs with compile-time K trip counts (16 / 17 / 17).
  proj_tiles<16, NI>(xnA, qWt, qB, qh, n, s0, wv, m, hi);
  proj_tiles<17, KPAD>(xpA, kWt, kB, kh, n, s0, wv, m, hi);
  proj_tiles<17, KPAD>(xpA, vWt, vB, vh, n, s0, wv, m, hi);
}

// ---------------------------------------------------------------------------
// Kernel 3: flash attention. grid = (n*8 heads) x (8 query chunks of 128).
// Each wave owns one 16-row query tile; K staged to LDS via async DMA,
// V staged transposed so the P.V B-fragment is two ds_load_b128.
// ---------------------------------------------------------------------------
__global__ __launch_bounds__(256) void attn_kernel(
    const _Float16* __restrict__ qh, const _Float16* __restrict__ kh,
    const _Float16* __restrict__ vh, _Float16* __restrict__ oh) {
  int nh = blockIdx.x >> 3;         // n*8 + h
  int qi = blockIdx.x & 7;
  int n = nh >> 3, h = nh & 7;
  int lane = threadIdx.x & 31, wv = threadIdx.x >> 5;
  int m = lane & 15, hi = lane >> 4;
  int s0 = (qi * 8 + wv) * 16;      // this wave's 16 query rows

  __shared__ _Float16 Kt[128][DH] __attribute__((aligned(16)));    // [t][d] 16 KB
  __shared__ _Float16 VtT[DH][128] __attribute__((aligned(16)));   // [d][t] 16 KB
  __shared__ _Float16 Pt[8][16][32] __attribute__((aligned(16)));  // 8 KB

  const _Float16* qp = qh + (size_t)nh * SEQ * DH;
  const _Float16* kp = kh + (size_t)nh * SEQ * DH;
  const _Float16* vp = vh + (size_t)nh * SEQ * DH;
  unsigned kt_lds = (unsigned)(uintptr_t)&Kt[0][0];   // LDS byte address

  // Q fragments (d = 0..31 and 32..63), loaded once.
  const _Float16* qrow = qp + (size_t)(s0 + m) * DH;
  v16h qa0 = load_afrag(qrow, 0, hi);
  v16h qa1 = load_afrag(qrow, 32, hi);

  float mrow[8], lrow[8];
  v8f oacc[4];
#pragma unroll
  for (int r = 0; r < 8; ++r) {
    mrow[r] = -1e30f;
    lrow[r] = 0.f;
    oacc[0][r] = oacc[1][r] = oacc[2][r] = oacc[3][r] = 0.f;
  }
  const float scale = 0.125f;       // 1/sqrt(512/8)

  for (int ch = 0; ch < 8; ++ch) {
    int t0 = ch * 128;
    __syncthreads();                // previous chunk fully consumed

    // K chunk: LDS-direct async copy (4 x b128 per lane, ASYNCcnt-tracked).
    const char* kg = (const char*)(kp + (size_t)t0 * DH);
    for (int i = threadIdx.x; i < 128 * DH / 8; i += 256)
      async_ld_b128(kt_lds + (unsigned)i * 16u, kg + (size_t)i * 16);

    // V chunk: load b128 and scatter transposed into VtT[d][t].
    const uint4* vg = (const uint4*)(vp + (size_t)t0 * DH);
    for (int i = threadIdx.x; i < 128 * DH / 8; i += 256) {
      int t = (i * 8) >> 6, d0 = (i * 8) & 63;
      union { uint4 q; _Float16 hh[8]; } u;
      u.q = vg[i];
#pragma unroll
      for (int e = 0; e < 8; ++e) VtT[d0 + e][t] = u.hh[e];
    }
    if (ch < 7) {                   // warm L2 for the next chunk
      __builtin_prefetch(kp + (size_t)(t0 + 128) * DH, 0, 1);
      __builtin_prefetch(vp + (size_t)(t0 + 128) * DH, 0, 1);
    }
    wait_async0();                  // this wave's async K stores visible
    __syncthreads();                // all waves' staging visible

    for (int sc = 0; sc < 4; ++sc) {          // 32-key subchunks
      int tt = sc * 32;
      // S = Q K^T for two 16-key subtiles (B: lane=key col, K-dim=d).
      v8f sA, sB;
#pragma unroll
      for (int r = 0; r < 8; ++r) { sA[r] = 0.f; sB[r] = 0.f; }
      sA = wmma32(qa0, load_bfrag(&Kt[tt + m][0], hi), sA);
      sA = wmma32(qa1, load_bfrag(&Kt[tt + m][32], hi), sA);
      sB = wmma32(qa0, load_bfrag(&Kt[tt + 16 + m][0], hi), sB);
      sB = wmma32(qa1, load_bfrag(&Kt[tt + 16 + m][32], hi), sB);

      // Online softmax across this 32-key chunk.
#pragma unroll
      for (int r = 0; r < 8; ++r) {
        float v0 = sA[r] * scale, v1 = sB[r] * scale;
        float mx = redmax16(fmaxf(v0, v1));
        float mnew = fmaxf(mrow[r], mx);
        float alpha = __expf(mrow[r] - mnew);
        mrow[r] = mnew;
        float p0 = __expf(v0 - mnew), p1 = __expf(v1 - mnew);
        lrow[r] = lrow[r] * alpha + redsum16(p0 + p1);
        oacc[0][r] *= alpha; oacc[1][r] *= alpha;
        oacc[2][r] *= alpha; oacc[3][r] *= alpha;
        int R = r + hi * 8;
        Pt[wv][R][m] = (_Float16)p0;        // C-layout -> LDS (row-major)
        Pt[wv][R][16 + m] = (_Float16)p1;
      }
      // Re-read P in the A-matrix layout (per-wave region, DS ops in order).
      v16h pa = load_afrag(&Pt[wv][m][0], 0, hi);

      // O += P V : B[K=key][N=d] = VtT[d][key], contiguous in key.
#pragma unroll
      for (int f = 0; f < 4; ++f) {
        v16h b = load_bfrag(&VtT[f * 16 + m][tt], hi);
        oacc[f] = wmma32(pa, b, oacc[f]);
      }
    }
  }

  // Normalize and store as f16 [n][s][c], c = h*64 + d.
#pragma unroll
  for (int r = 0; r < 8; ++r) {
    float inv = 1.f / lrow[r];
    int R = r + hi * 8;
    _Float16* orow = oh + ((size_t)n * SEQ + s0 + R) * NI + h * DH;
#pragma unroll
    for (int f = 0; f < 4; ++f)
      orow[f * 16 + m] = (_Float16)(oacc[f][r] * inv);
  }
}

// ---------------------------------------------------------------------------
// Kernel 4: out[n][c][s] = (o @ proj_W + proj_b) transposed, fp32.
// ---------------------------------------------------------------------------
__global__ __launch_bounds__(256) void proj_kernel(
    const _Float16* __restrict__ oh, const _Float16* __restrict__ pWt,
    const float* __restrict__ pB, float* __restrict__ out) {
  int n = blockIdx.x >> 6;
  int s0 = (blockIdx.x & 63) * 16;
  int lane = threadIdx.x & 31, wv = threadIdx.x >> 5;
  int m = lane & 15, hi = lane >> 4;
  const _Float16* orow = oh + ((size_t)n * SEQ + s0 + m) * NI;

  for (int tile = wv; tile < 32; tile += 8) {
    int c0 = tile * 16, col = c0 + m;
    const _Float16* wcol = pWt + (size_t)col * NI;
    v8f acc;
#pragma unroll
    for (int r = 0; r < 8; ++r) acc[r] = 0.f;
#pragma unroll
    for (int kc = 0; kc < 16; ++kc) {
      int k0 = kc * 32;
      v16h a = load_afrag(orow, k0, hi);
      v16h b = load_bfrag(wcol + k0, hi);
      acc = wmma32(a, b, acc);
    }
    float bias = pB[col];
#pragma unroll
    for (int r = 0; r < 8; ++r) {
      int R = r + hi * 8;
      out[((size_t)n * NI + col) * SEQ + s0 + R] = acc[r] + bias;
    }
  }
}

// ---------------------------------------------------------------------------
extern "C" void kernel_launch(void* const* d_in, const int* in_sizes, int n_in,
                              void* d_out, int out_size, void* d_ws,
                              size_t ws_size, hipStream_t stream) {
  const float* x     = (const float*)d_in[0];
  const float* pose  = (const float*)d_in[1];
  const float* ng    = (const float*)d_in[2];
  const float* nb    = (const float*)d_in[3];
  const float* png   = (const float*)d_in[4];
  const float* pnb   = (const float*)d_in[5];
  const float* poseW = (const float*)d_in[6];
  const float* poseB = (const float*)d_in[7];
  const float* qW    = (const float*)d_in[8];
  const float* qB    = (const float*)d_in[9];
  const float* kW    = (const float*)d_in[10];
  const float* kB    = (const float*)d_in[11];
  const float* vW    = (const float*)d_in[12];
  const float* vB    = (const float*)d_in[13];
  const float* pW    = (const float*)d_in[14];
  const float* pB    = (const float*)d_in[15];
  float* out = (float*)d_out;

  char* ws = (char*)d_ws;
  float* pe_t = (float*)ws;                                   // 64 KB
  _Float16* qWt = (_Float16*)(ws + (64 << 10));               // 512 KB
  _Float16* kWt = qWt + (size_t)NI * NI;                      // 544 KB
  _Float16* vWt = kWt + (size_t)NI * KPAD;                    // 544 KB
  _Float16* pWt = vWt + (size_t)NI * KPAD;                    // 512 KB
  _Float16* qh  = (_Float16*)(ws + ((size_t)4 << 20));        // 8 MB each
  const size_t HBUF = (size_t)8 * NHD * SEQ * DH;
  _Float16* kh = qh + HBUF;
  _Float16* vh = kh + HBUF;
  _Float16* oh = vh + HBUF;
  (void)in_sizes; (void)n_in; (void)out_size; (void)ws_size;

  wprep_kernel<<<(NI * NI) / 256, 256, 0, stream>>>(qW, qWt, NI, NI);
  wprep_kernel<<<(NI * KPAD) / 256, 256, 0, stream>>>(kW, kWt, KDIM2, KPAD);
  wprep_kernel<<<(NI * KPAD) / 256, 256, 0, stream>>>(vW, vWt, KDIM2, KPAD);
  wprep_kernel<<<(NI * NI) / 256, 256, 0, stream>>>(pW, pWt, NI, NI);
  pose_kernel<<<32, 256, 0, stream>>>(pose, poseW, poseB, pe_t);
  qkv_kernel<<<512, 256, 0, stream>>>(x, pe_t, ng, nb, png, pnb,
                                      qWt, qB, kWt, kB, vWt, vB, qh, kh, vh);
  attn_kernel<<<512, 256, 0, stream>>>(qh, kh, vh, oh);
  proj_kernel<<<512, 256, 0, stream>>>(oh, pWt, pB, out);
}
